// SpikingNetwork_45337674776884
// MI455X (gfx1250) — matrix operand order repaired and checked by reading, not compiled, not run on hardware
//
#include <hip/hip_runtime.h>
#include <hip/hip_bf16.h>
#include <stdint.h>

// ---- CDNA5 (gfx1250) wave32 WMMA types ----
typedef __bf16 bf16_t;
typedef bf16_t v16bf __attribute__((ext_vector_type(16)));
typedef bf16_t v8bf  __attribute__((ext_vector_type(8)));
typedef float  v8f   __attribute__((ext_vector_type(8)));

// ---- problem sizes (fixed by reference) ----
#define B_    512
#define IN_   3072
#define HID_  2048
#define OUT_  512
#define T_    10
#define DECAY_ 0.3f
#define VTH_   1.0f

// Counter-based hash RNG -> U[0,1). Deterministic per (t, idx).
__device__ __forceinline__ float hash_u01(uint32_t t, uint32_t i) {
  uint32_t x = i ^ (t * 0x9E3779B9u) ^ 0x2545F491u;
  x ^= x >> 16; x *= 0x7FEB352Du;
  x ^= x >> 15; x *= 0x846CA68Bu;
  x ^= x >> 16;
  return (float)(x >> 8) * (1.0f / 16777216.0f);
}

// ---------------- prep kernels ----------------
__global__ void k_sigmoid(const float* __restrict__ x, float* __restrict__ p, int n) {
  int i = blockIdx.x * blockDim.x + threadIdx.x;
  if (i < n) p[i] = 1.0f / (1.0f + __expf(-x[i]));
}

__global__ void k_cvt_bf16(const float* __restrict__ w, bf16_t* __restrict__ o, int n) {
  int i = blockIdx.x * blockDim.x + threadIdx.x;
  if (i < n) o[i] = (bf16_t)w[i];
}

__global__ void k_zero(float* __restrict__ v, float* __restrict__ s) {
  int i = blockIdx.x * blockDim.x + threadIdx.x;
  const int n1 = B_ * HID_;
  const int tot = n1 + B_ * OUT_;
  if (i < tot) {
    if (i < n1) v[i] = 0.0f;
    else        s[i - n1] = 0.0f;
  }
}

__global__ void k_spikes(const float* __restrict__ p, bf16_t* __restrict__ pre,
                         int t, int n) {
  int i = blockIdx.x * blockDim.x + threadIdx.x;
  if (i < n) pre[i] = (hash_u01((uint32_t)t, (uint32_t)i) < p[i]) ? (bf16_t)1.0f
                                                                  : (bf16_t)0.0f;
}

// ---------------- WMMA fragment loader (GVS addressing) ----------------
// base is wave-uniform (lives in SGPRs); off is a loop-invariant 32-bit per-lane
// element offset; hi_delta / any chunk delta folds into the load's IOFFSET.
// -> global_load_b128 vdst, v_off, s[base]  with zero per-iteration VALU addr math.
__device__ __forceinline__ v16bf load_frag16(const bf16_t* __restrict__ base,
                                             int off, int hi_delta) {
  v8bf lo = *(const v8bf*)(base + off);
  v8bf hi = *(const v8bf*)(base + off + hi_delta);
  return __builtin_shufflevector(lo, hi, 0,1,2,3,4,5,6,7,8,9,10,11,12,13,14,15);
}

#define WMMA_BF16(ACC, AF, BF)                                           \
  (ACC) = __builtin_amdgcn_wmma_f32_16x16x32_bf16(                       \
      false, (AF), false, (BF), (short)0, (ACC), false, false)

// Double-buffered NT GEMM core: wave tile 64(M) x 32(N) = 4x2 WMMA tiles.
// A (16x32 MxK) frag: lanes 0-15 M=lane, K 0..15;
//                     lanes 16-31 same M, K 16..31.       (2x b128, contiguous)
// B (32x16 KxN) frag from row-major W[N][K]: lanes 0-15 col N=lane, K 0..15;
//                     lanes 16-31 col N=lane-16, K 16..31. (2x b128, contiguous)
// K unrolled by 64; fragments for k+32/k+64 in flight during WMMAs on k/k+32.
template <int K>
__device__ __forceinline__ void gemm_nt_64x32(const bf16_t* __restrict__ A,
                                              const bf16_t* __restrict__ W,
                                              int m0, int n0, int lane,
                                              v8f acc[4][2]) {
  static_assert(K % 64 == 0, "K must be a multiple of 64");
  const int ml  = lane & 15;
  const int kha = (lane >> 4) ? 16 : 0;   // A: K-half select
  const int khb = (lane >> 4) ? 16 : 0;   // B: K-half select
  int aoff[4], boff[2];
#pragma unroll
  for (int i = 0; i < 4; ++i) aoff[i] = (m0 + 16 * i + ml) * K + kha;
#pragma unroll
  for (int j = 0; j < 2; ++j) boff[j] = (n0 + 16 * j + ml) * K + khb;

  const bf16_t* Ak = A;   // uniform, advances 64 elements/iter (SALU add)
  const bf16_t* Wk = W;
  v16bf a[4], b[2], a2[4], b2[2];

#pragma unroll
  for (int i = 0; i < 4; ++i) a[i] = load_frag16(Ak, aoff[i], 8);
#pragma unroll
  for (int j = 0; j < 2; ++j) b[j] = load_frag16(Wk, boff[j], 8);

#pragma unroll 1
  for (int k0 = 0; k0 < K - 64; k0 += 64) {
    // stage chunk k0+32 while computing chunk k0
#pragma unroll
    for (int i = 0; i < 4; ++i) a2[i] = load_frag16(Ak, aoff[i] + 32, 8);
#pragma unroll
    for (int j = 0; j < 2; ++j) b2[j] = load_frag16(Wk, boff[j] + 32, 8);
#pragma unroll
    for (int i = 0; i < 4; ++i)
#pragma unroll
      for (int j = 0; j < 2; ++j) WMMA_BF16(acc[i][j], a[i], b[j]);

    // stage chunk k0+64 while computing chunk k0+32
#pragma unroll
    for (int i = 0; i < 4; ++i) a[i] = load_frag16(Ak, aoff[i] + 64, 8);
#pragma unroll
    for (int j = 0; j < 2; ++j) b[j] = load_frag16(Wk, boff[j] + 64, 8);
#pragma unroll
    for (int i = 0; i < 4; ++i)
#pragma unroll
      for (int j = 0; j < 2; ++j) WMMA_BF16(acc[i][j], a2[i], b2[j]);

    Ak += 64;
    Wk += 64;
  }

  // peeled tail: chunks K-64 (already in a/b) and K-32
#pragma unroll
  for (int i = 0; i < 4; ++i) a2[i] = load_frag16(Ak, aoff[i] + 32, 8);
#pragma unroll
  for (int j = 0; j < 2; ++j) b2[j] = load_frag16(Wk, boff[j] + 32, 8);
#pragma unroll
  for (int i = 0; i < 4; ++i)
#pragma unroll
    for (int j = 0; j < 2; ++j) WMMA_BF16(acc[i][j], a[i], b[j]);
#pragma unroll
  for (int i = 0; i < 4; ++i)
#pragma unroll
    for (int j = 0; j < 2; ++j) WMMA_BF16(acc[i][j], a2[i], b2[j]);
}

// Block: 256 thr = 8 waves (2x4) -> 128x128 workgroup tile. Dims divide evenly.

// GEMM1 + LIF:  h = pre @ W_in^T ;  v = v*DECAY + h ; fire/reset ; spk out (bf16)
__global__ __launch_bounds__(256)
void k_gemm1_lif(const bf16_t* __restrict__ pre,  // [B_, IN_]
                 const bf16_t* __restrict__ Wi,   // [HID_, IN_]
                 float* __restrict__ v,           // [B_, HID_]
                 bf16_t* __restrict__ spk)        // [B_, HID_]
{
  const int lane = threadIdx.x & 31;
  const int w    = threadIdx.x >> 5;
  const int m0   = blockIdx.x * 128 + (w >> 2) * 64;
  const int n0   = blockIdx.y * 128 + (w & 3) * 32;

  v8f acc[4][2];
#pragma unroll
  for (int i = 0; i < 4; ++i)
#pragma unroll
    for (int j = 0; j < 2; ++j) acc[i][j] = (v8f)0.0f;

  gemm_nt_64x32<IN_>(pre, Wi, m0, n0, lane, acc);

  // Epilogue: C/D layout -> element e of VGPR block is (M = 8*(lane>=16)+e, N = lane&15)
  const int nl = lane & 15;
  const int mh = (lane >> 4) ? 8 : 0;
#pragma unroll
  for (int i = 0; i < 4; ++i)
#pragma unroll
    for (int j = 0; j < 2; ++j) {
      const int n = n0 + 16 * j + nl;
#pragma unroll
      for (int e = 0; e < 8; ++e) {
        const int m = m0 + 16 * i + mh + e;
        const size_t idx = (size_t)m * HID_ + n;
        float vv = v[idx] * DECAY_ + acc[i][j][e];
        const bool fire = (vv >= VTH_);
        v[idx]   = fire ? 0.0f : vv;
        spk[idx] = fire ? (bf16_t)1.0f : (bf16_t)0.0f;
      }
    }
}

// GEMM2 + count:  out = spk @ W_out^T + b ;  s += (out > 0)
__global__ __launch_bounds__(256)
void k_gemm2_count(const bf16_t* __restrict__ spk,  // [B_, HID_]
                   const bf16_t* __restrict__ Wo,   // [OUT_, HID_]
                   const float* __restrict__ bias,  // [OUT_]
                   float* __restrict__ s)           // [B_, OUT_]
{
  const int lane = threadIdx.x & 31;
  const int w    = threadIdx.x >> 5;
  const int m0   = blockIdx.x * 128 + (w >> 2) * 64;
  const int n0   = blockIdx.y * 128 + (w & 3) * 32;

  v8f acc[4][2];
#pragma unroll
  for (int i = 0; i < 4; ++i)
#pragma unroll
    for (int j = 0; j < 2; ++j) acc[i][j] = (v8f)0.0f;

  gemm_nt_64x32<HID_>(spk, Wo, m0, n0, lane, acc);

  const int nl = lane & 15;
  const int mh = (lane >> 4) ? 8 : 0;
#pragma unroll
  for (int i = 0; i < 4; ++i)
#pragma unroll
    for (int j = 0; j < 2; ++j) {
      const int n = n0 + 16 * j + nl;
      const float bn = bias[n];
#pragma unroll
      for (int e = 0; e < 8; ++e) {
        const int m = m0 + 16 * i + mh + e;
        const size_t idx = (size_t)m * OUT_ + n;
        const float outv = acc[i][j][e] + bn;
        if (outv > 0.0f) s[idx] = s[idx] + 1.0f;  // unique owner: no atomic needed
      }
    }
}

// ---------------- launcher ----------------
extern "C" void kernel_launch(void* const* d_in, const int* in_sizes, int n_in,
                              void* d_out, int out_size, void* d_ws, size_t ws_size,
                              hipStream_t stream) {
  (void)in_sizes; (void)n_in; (void)out_size; (void)ws_size;
  const float* x     = (const float*)d_in[0];   // [512,3,32,32]
  const float* W_in  = (const float*)d_in[1];   // [2048,3072]
  const float* W_out = (const float*)d_in[2];   // [512,2048]
  const float* b_out = (const float*)d_in[3];   // [512]
  // d_in[4] is T on device; host loop count is the reference constant T_=10.
  float* s = (float*)d_out;                      // [512,512]

  // Carve workspace (all 256B aligned): ~30.4 MB total, fits in 192 MB L2.
  char* ws = (char*)d_ws;
  size_t off = 0;
  auto carve = [&](size_t bytes) -> void* {
    off = (off + 255) & ~(size_t)255;
    void* q = ws + off;
    off += bytes;
    return q;
  };
  float*  p    = (float*) carve((size_t)B_ * IN_ * 4);    // sigmoid probs
  bf16_t* pre  = (bf16_t*)carve((size_t)B_ * IN_ * 2);    // input spikes
  bf16_t* Wib  = (bf16_t*)carve((size_t)HID_ * IN_ * 2);  // W_in bf16
  bf16_t* Wob  = (bf16_t*)carve((size_t)OUT_ * HID_ * 2); // W_out bf16
  float*  v    = (float*) carve((size_t)B_ * HID_ * 4);   // membrane
  bf16_t* spk  = (bf16_t*)carve((size_t)B_ * HID_ * 2);   // hidden spikes

  const int nP  = B_ * IN_;
  const int nWi = HID_ * IN_;
  const int nWo = OUT_ * HID_;
  const int nZ  = B_ * HID_ + B_ * OUT_;

  k_sigmoid <<<(nP  + 255) / 256, 256, 0, stream>>>(x, p, nP);
  k_cvt_bf16<<<(nWi + 255) / 256, 256, 0, stream>>>(W_in, Wib, nWi);
  k_cvt_bf16<<<(nWo + 255) / 256, 256, 0, stream>>>(W_out, Wob, nWo);
  k_zero    <<<(nZ  + 255) / 256, 256, 0, stream>>>(v, s);

  for (int t = 0; t < T_; ++t) {
    k_spikes<<<(nP + 255) / 256, 256, 0, stream>>>(p, pre, t, nP);
    k_gemm1_lif  <<<dim3(B_ / 128, HID_ / 128), 256, 0, stream>>>(pre, Wib, v, spk);
    k_gemm2_count<<<dim3(B_ / 128, OUT_ / 128), 256, 0, stream>>>(spk, Wob, b_out, s);
  }
}
